// DynamicFeatureGroupingLayer_49331994362158
// MI455X (gfx1250) — matrix operand order repaired and checked by reading, not compiled, not run on hardware
//
#include <hip/hip_runtime.h>
#include <hip/hip_bf16.h>
#include <math.h>

typedef __attribute__((ext_vector_type(16))) _Float16 v16h;
typedef __attribute__((ext_vector_type(8)))  _Float16 v8h;
typedef __attribute__((ext_vector_type(8)))  float    v8f;

#define B_TOT   4096
#define G_TOT   10
#define M_TOT   4
#define S_TOT   100
#define H_TOT   512
#define KPAD    128
#define OUTW    (G_TOT * H_TOT)   // 5120

// ---------------------------------------------------------------------------
// Prep 1: entmax(1.1) mask over W_masks rows. 40 rows of 100. One block/row.
// ---------------------------------------------------------------------------
__global__ void entmax_mask_kernel(const float* __restrict__ Wm,
                                   float* __restrict__ mask) {
    int row = blockIdx.x;                 // (g*4 + m), 0..39
    int t   = threadIdx.x;                // 128 threads
    const float* v = Wm + row * S_TOT;
    __shared__ float red[128];

    float val = (t < S_TOT) ? v[t] : -INFINITY;
    red[t] = val; __syncthreads();
    for (int s = 64; s > 0; s >>= 1) {
        if (t < s) red[t] = fmaxf(red[t], red[t + s]);
        __syncthreads();
    }
    float mx = red[0]; __syncthreads();

    float e  = (t < S_TOT) ? expf(val - mx) : 0.f;
    float ep = (t < S_TOT) ? powf(e, 1.1f)  : 0.f;
    red[t] = ep; __syncthreads();
    for (int s = 64; s > 0; s >>= 1) {
        if (t < s) red[t] += red[t + s];
        __syncthreads();
    }
    float denom = powf(red[0] + 1e-5f, 1.0f / 1.1f);
    if (t < S_TOT) mask[row * S_TOT + t] = e / denom;
}

// ---------------------------------------------------------------------------
// Prep 2: W [G,H,100] f32 -> [G,H,128] f16, zero padded K.
// ---------------------------------------------------------------------------
__global__ void convW_kernel(const float* __restrict__ W,
                             _Float16* __restrict__ Wh, int total) {
    int i = blockIdx.x * 256 + threadIdx.x;
    if (i >= total) return;
    int k  = i & (KPAD - 1);
    int hs = i >> 7;                       // g*512 + h
    Wh[i] = (k < S_TOT) ? (_Float16)W[hs * S_TOT + k] : (_Float16)0.f;
}

// ---------------------------------------------------------------------------
// Fused: (x*mask) GEMM W1 & W2 (f16 WMMA, f32 acc) -> LN -> sigmoid*relu ->
// sum over masks -> out. Block = (g, 8 batch rows), 8 waves, full H=512.
// ---------------------------------------------------------------------------
__global__ __launch_bounds__(256)
void fused_dfg_kernel(const float* __restrict__ x,
                      const _Float16* __restrict__ W1h,
                      const _Float16* __restrict__ W2h,
                      const float* __restrict__ maskf,
                      const float* __restrict__ ln1w, const float* __restrict__ ln1b,
                      const float* __restrict__ ln2w, const float* __restrict__ ln2b,
                      float* __restrict__ out) {
    const int g      = blockIdx.y;
    const int b_base = blockIdx.x * 8;

    __shared__ _Float16 As[32 * KPAD];          // 8 KB weighted tile, rows r=b*4+m
    __shared__ float    partial[8][2][32][2];   // [wave][mat][row][sum,sumsq]
    __shared__ float    stats[32][2][2];        // [row][mat][mean,rstd]

    const int tid   = threadIdx.x;
    const int lane  = tid & 31;
    const int wave  = tid >> 5;
    const int nlane = lane & 15;
    const int hi    = lane >> 4;

    // ---- build A = x * entmax(mask), f16, K padded to 128 ----
    const float* xg = x + (size_t)b_base * 1000 + g * S_TOT;
    const float* mg = maskf + g * (M_TOT * S_TOT);
    for (int idx = tid; idx < 32 * KPAD; idx += 256) {
        int r = idx >> 7, k = idx & (KPAD - 1);
        float v = 0.f;
        if (k < S_TOT) v = xg[(r >> 2) * 1000 + k] * mg[(r & 3) * S_TOT + k];
        As[idx] = (_Float16)v;
    }
    __syncthreads();

    // ---- GEMM: 32 rows x 512 cols, K=128. Wave w owns cols [w*64, w*64+64) ----
    v8f acc[2][2][4] = {};                      // [mat][rowtile][coltile]
    const int nbase = wave * 64;
    const _Float16* W1g = W1h + (size_t)g * H_TOT * KPAD;
    const _Float16* W2g = W2h + (size_t)g * H_TOT * KPAD;

    #pragma unroll
    for (int ks = 0; ks < 4; ++ks) {
        // A fragments (ISA 16-bit A layout: lane-half K chunks 8+8)
        const int ka = ks * 32 + hi * 8;
        v16h afrag[2];
        #pragma unroll
        for (int t = 0; t < 2; ++t) {
            const _Float16* p = &As[(t * 16 + nlane) * KPAD + ka];
            union { v16h v; v8h h[2]; } u;
            u.h[0] = *(const v8h*)(p);
            u.h[1] = *(const v8h*)(p + 16);
            afrag[t] = u.v;
        }
        // B fragments (same K layout as A: lane-half K chunks 8+8)
        const int kb = ks * 32 + hi * 8;
        #pragma unroll
        for (int mat = 0; mat < 2; ++mat) {
            const _Float16* Wg = mat ? W2g : W1g;
            #pragma unroll
            for (int c = 0; c < 4; ++c) {
                const _Float16* p = Wg + (size_t)(nbase + c * 16 + nlane) * KPAD + kb;
                union { v16h v; v8h h[2]; } u;
                u.h[0] = *(const v8h*)(p);
                u.h[1] = *(const v8h*)(p + 16);
                v16h bfrag = u.v;
                #pragma unroll
                for (int t = 0; t < 2; ++t)
                    acc[mat][t][c] = __builtin_amdgcn_wmma_f32_16x16x32_f16(
                        false, afrag[t], false, bfrag,
                        (short)0, acc[mat][t][c], false, false);
            }
        }
    }

    // ---- LayerNorm partial sums: per-lane over 4 col-tiles, then 16-lane
    //      butterfly within each lane half (rows differ between halves) ----
    #pragma unroll
    for (int mat = 0; mat < 2; ++mat)
        #pragma unroll
        for (int t = 0; t < 2; ++t)
            #pragma unroll
            for (int j = 0; j < 8; ++j) {
                float s = 0.f, q = 0.f;
                #pragma unroll
                for (int c = 0; c < 4; ++c) {
                    float v = acc[mat][t][c][j];
                    s += v; q += v * v;
                }
                #pragma unroll
                for (int o = 1; o < 16; o <<= 1) {
                    s += __shfl_xor(s, o, 32);
                    q += __shfl_xor(q, o, 32);
                }
                int row = t * 16 + j + hi * 8;
                if (nlane == 0) {
                    partial[wave][mat][row][0] = s;
                    partial[wave][mat][row][1] = q;
                }
            }
    __syncthreads();

    if (tid < 64) {
        int mat = tid & 1, row = tid >> 1;
        float s = 0.f, q = 0.f;
        #pragma unroll
        for (int w = 0; w < 8; ++w) { s += partial[w][mat][row][0]; q += partial[w][mat][row][1]; }
        float mean = s * (1.f / (float)H_TOT);
        float var  = q * (1.f / (float)H_TOT) - mean * mean;
        stats[row][mat][0] = mean;
        stats[row][mat][1] = rsqrtf(var + 1e-5f);
    }
    __syncthreads();

    // ---- LN + sigmoid(n1)*n2, relu, sum over 4 masks (in-lane, VGPR groups
    //      of 4 because rows are ordered r = b*4 + m), store ----
    const float* l1w = ln1w + g * H_TOT; const float* l1b = ln1b + g * H_TOT;
    const float* l2w = ln2w + g * H_TOT; const float* l2b = ln2b + g * H_TOT;
    #pragma unroll
    for (int t = 0; t < 2; ++t) {
        #pragma unroll
        for (int c = 0; c < 4; ++c) {
            int col = nbase + c * 16 + nlane;
            float w1c = l1w[col], b1c = l1b[col];
            float w2c = l2w[col], b2c = l2b[col];
            float o0 = 0.f, o1 = 0.f;
            #pragma unroll
            for (int j = 0; j < 8; ++j) {
                int row = t * 16 + j + hi * 8;
                float n1 = (acc[0][t][c][j] - stats[row][0][0]) * stats[row][0][1] * w1c + b1c;
                float n2 = (acc[1][t][c][j] - stats[row][1][0]) * stats[row][1][1] * w2c + b2c;
                float sg = 1.f / (1.f + expf(-n1));
                float a  = fmaxf(sg * n2, 0.f);
                if (j < 4) o0 += a; else o1 += a;
            }
            int b0 = t * 4 + hi * 2;           // batch row for VGPRs 0..3
            out[(size_t)(b_base + b0)     * OUTW + g * H_TOT + col] = o0;
            out[(size_t)(b_base + b0 + 1) * OUTW + g * H_TOT + col] = o1;
        }
    }
}

// ---------------------------------------------------------------------------
extern "C" void kernel_launch(void* const* d_in, const int* in_sizes, int n_in,
                              void* d_out, int out_size, void* d_ws, size_t ws_size,
                              hipStream_t stream) {
    const float* x     = (const float*)d_in[0];
    const float* Wm    = (const float*)d_in[1];
    const float* W1    = (const float*)d_in[2];
    const float* W2    = (const float*)d_in[3];
    const float* ln1w  = (const float*)d_in[4];
    const float* ln1b  = (const float*)d_in[5];
    const float* ln2w  = (const float*)d_in[6];
    const float* ln2b  = (const float*)d_in[7];
    float* out = (float*)d_out;

    const int wElems = G_TOT * H_TOT * KPAD;        // 655360 per matrix
    _Float16* W1h   = (_Float16*)d_ws;
    _Float16* W2h   = W1h + wElems;
    float*    maskf = (float*)((char*)d_ws + (size_t)2 * wElems * sizeof(_Float16));

    convW_kernel<<<(wElems + 255) / 256, 256, 0, stream>>>(W1, W1h, wElems);
    convW_kernel<<<(wElems + 255) / 256, 256, 0, stream>>>(W2, W2h, wElems);
    entmax_mask_kernel<<<G_TOT * M_TOT, 128, 0, stream>>>(Wm, maskf);

    dim3 grid(B_TOT / 8, G_TOT);                    // 512 x 10 blocks
    fused_dfg_kernel<<<grid, 256, 0, stream>>>(x, W1h, W2h, maskf,
                                               ln1w, ln1b, ln2w, ln2b, out);
}
